// MultiHeadAttention_30468497998628
// MI455X (gfx1250) — matrix operand order repaired and checked by reading, not compiled, not run on hardware
//
#include <hip/hip_runtime.h>
#include <hip/hip_bf16.h>

// ---------------------------------------------------------------------------
// MHA forward for MI455X (gfx1250): f16 WMMA GEMMs with async LDS-staged B
// tiles, flash attention with 64-wide k blocks.  DIM=1024, H=16, Dh=64,
// B=2, T=2048.
// ---------------------------------------------------------------------------

#define DIM   1024
#define NH    16
#define DH    64
#define BBATCH 2
#define TT    2048
#define MTOT  (BBATCH * TT)      // 4096 rows

typedef __attribute__((ext_vector_type(16))) _Float16 v16h;
typedef __attribute__((ext_vector_type(8)))  _Float16 v8h;
typedef __attribute__((ext_vector_type(8)))  float    v8f;
typedef __attribute__((ext_vector_type(4)))  int      v4i;

union U16H { v16h v; v8h h[2]; };

static __device__ __forceinline__ v8f wmma_f16(v16h a, v16h b, v8f c) {
    return __builtin_amdgcn_wmma_f32_16x16x32_f16(false, a, false, b, (short)0, c,
                                                  false, false);
}

// ------------------------- async copy (gfx1250) ----------------------------
#if __has_builtin(__builtin_amdgcn_global_load_async_to_lds_b128)
#define HAVE_ASYNC 1
#endif

static __device__ __forceinline__ void async_cp16(const _Float16* g, _Float16* l) {
#ifdef HAVE_ASYNC
    typedef __attribute__((address_space(1))) v4i* gptr_t;   // global b128
    typedef __attribute__((address_space(3))) v4i* lptr_t;   // LDS b128
    __builtin_amdgcn_global_load_async_to_lds_b128((gptr_t)g, (lptr_t)l, 0, 0);
#else
    *(v8h*)l = *(const v8h*)g;      // fallback: stage through VGPRs
#endif
}

static __device__ __forceinline__ void async_wait_two() {
#ifdef HAVE_ASYNC
#if __has_builtin(__builtin_amdgcn_s_wait_asynccnt)
    __builtin_amdgcn_s_wait_asynccnt(2);
#else
    asm volatile("s_wait_asynccnt 0x2" ::: "memory");
#endif
#endif
}
static __device__ __forceinline__ void async_wait_zero() {
#ifdef HAVE_ASYNC
#if __has_builtin(__builtin_amdgcn_s_wait_asynccnt)
    __builtin_amdgcn_s_wait_asynccnt(0);
#else
    asm volatile("s_wait_asynccnt 0x0" ::: "memory");
#endif
#endif
}

// -------------------------- precision prep kernels -------------------------

__global__ void mha_cvt_f16(const float* __restrict__ src, _Float16* __restrict__ dst, int n) {
    for (int i = blockIdx.x * blockDim.x + threadIdx.x; i < n; i += gridDim.x * blockDim.x)
        dst[i] = (_Float16)src[i];
}

// src[rows][cols] (fp32) -> dst[cols][rows] (f16)
__global__ void mha_cvt_t_f16(const float* __restrict__ src, _Float16* __restrict__ dst,
                              int rows, int cols) {
    int n = rows * cols;
    for (int i = blockIdx.x * blockDim.x + threadIdx.x; i < n; i += gridDim.x * blockDim.x) {
        int r = i / cols, c = i - r * cols;
        dst[(size_t)c * rows + r] = (_Float16)src[i];
    }
}

// ----------------------------- GEMM mainloop -------------------------------
// Block = 128 threads (4 waves).  Block tile 128(M) x 64(N); wave tile 32x64.
// B tile (64 rows x 32 halves) staged in LDS, double buffered, async copies.
#define BROW 40    // padded LDS row stride (halves): conflict-free b128 reads

// one thread stages two 16B chunks of the 4KB B tile
static __device__ __forceinline__ void stage_btile(const _Float16* __restrict__ Wt,
                                                   _Float16* lbuf, int n0, int k0, int tid) {
    int c   = tid << 1;          // chunk id (even)
    int row = c >> 2;            // 0..63
    int off = (c & 3) * 8;       // halves within row (0 or 16)
    const _Float16* g = Wt + (size_t)(n0 + row) * DIM + k0 + off;
    _Float16*       l = lbuf + row * BROW + off;
    async_cp16(g, l);
    async_cp16(g + 8, l + 8);
}

// mainloop shared by both GEMMs; acc[2][4] = two 16-row groups x four n tiles
static __device__ __forceinline__ void gemm_mainloop(const _Float16* __restrict__ A,
                                                     const _Float16* __restrict__ Wt,
                                                     _Float16* bsm /*2 bufs*/,
                                                     int m0, int n0, int tid,
                                                     int hl, int lr, v8f acc[2][4]) {
    const _Float16* arow0 = A + (size_t)(m0 + lr) * DIM;
    const _Float16* arow1 = A + (size_t)(m0 + 16 + lr) * DIM;

    stage_btile(Wt, bsm, n0, 0, tid);
    for (int k0 = 0; k0 < DIM; k0 += 32) {
        const int buf = (k0 >> 5) & 1;
        _Float16* cur = bsm + buf * (64 * BROW);
        if (k0 + 32 < DIM) {
            stage_btile(Wt, bsm + (buf ^ 1) * (64 * BROW), n0, k0 + 32, tid);
            async_wait_two();
        } else {
            async_wait_zero();
        }
        __syncthreads();

        U16H a0, a1;
        a0.h[0] = *(const v8h*)(arow0 + k0 + 8 * hl);
        a0.h[1] = *(const v8h*)(arow0 + k0 + 16 + 8 * hl);
        a1.h[0] = *(const v8h*)(arow1 + k0 + 8 * hl);
        a1.h[1] = *(const v8h*)(arow1 + k0 + 16 + 8 * hl);
#pragma unroll
        for (int j = 0; j < 4; ++j) {
            const _Float16* lb = cur + (16 * j + lr) * BROW + 8 * hl;
            U16H b;
            b.h[0] = *(const v8h*)(lb);
            b.h[1] = *(const v8h*)(lb + 16);
            acc[0][j] = wmma_f16(a0.v, b.v, acc[0][j]);
            acc[1][j] = wmma_f16(a1.v, b.v, acc[1][j]);
        }
        __syncthreads();
    }
}

// ------------------------------- QKV GEMM ----------------------------------
__global__ void __launch_bounds__(128)
mha_qkv_gemm(const _Float16* __restrict__ X, const _Float16* __restrict__ Wt,
             const float* __restrict__ bias,
             _Float16* __restrict__ Qf, _Float16* __restrict__ Kf, _Float16* __restrict__ Vtf) {
    __shared__ __align__(16) _Float16 bsm[2][64 * BROW];
    const int tid = threadIdx.x, wave = tid >> 5, lane = tid & 31;
    const int hl = lane >> 4, lr = lane & 15;
    const int m0 = blockIdx.x * 128 + wave * 32;
    const int n0 = blockIdx.y * 64;

    v8f acc[2][4] = {};
    gemm_mainloop(X, Wt, &bsm[0][0], m0, n0, tid, hl, lr, acc);

#pragma unroll
    for (int g = 0; g < 2; ++g) {
#pragma unroll
        for (int j = 0; j < 4; ++j) {
#pragma unroll
            for (int r = 0; r < 8; ++r) {
                int m = m0 + 16 * g + r + 8 * hl;
                int n = n0 + 16 * j + lr;
                float v = acc[g][j][r] + bias[n];
                int part = n >> 10;          // 0=Q 1=K 2=V
                int cc   = n & 1023;
                int hh   = cc >> 6;
                int d    = cc & 63;
                int bb   = m >> 11;          // T = 2048
                int t    = m & 2047;
                size_t bh = (size_t)bb * NH + hh;
                if (part == 0)      Qf [(bh * TT + t) * DH + d] = (_Float16)(v * 0.125f);
                else if (part == 1) Kf [(bh * TT + t) * DH + d] = (_Float16)v;
                else                Vtf[(bh * DH + d) * TT + t] = (_Float16)v;
            }
        }
    }
}

// ---------------------------- projection GEMM ------------------------------
__global__ void __launch_bounds__(128)
mha_proj_gemm(const _Float16* __restrict__ A, const _Float16* __restrict__ Wt,
              const float* __restrict__ bias, float* __restrict__ out) {
    __shared__ __align__(16) _Float16 bsm[2][64 * BROW];
    const int tid = threadIdx.x, wave = tid >> 5, lane = tid & 31;
    const int hl = lane >> 4, lr = lane & 15;
    const int m0 = blockIdx.x * 128 + wave * 32;
    const int n0 = blockIdx.y * 64;

    v8f acc[2][4] = {};
    gemm_mainloop(A, Wt, &bsm[0][0], m0, n0, tid, hl, lr, acc);

#pragma unroll
    for (int g = 0; g < 2; ++g) {
#pragma unroll
        for (int j = 0; j < 4; ++j) {
#pragma unroll
            for (int r = 0; r < 8; ++r) {
                int m = m0 + 16 * g + r + 8 * hl;
                int n = n0 + 16 * j + lr;
                out[(size_t)m * DIM + n] = acc[g][j][r] + bias[n];
            }
        }
    }
}

// ---------------------------- flash attention ------------------------------
// grid = (B*H, T/64), block = 128 (4 waves, 16 q-rows each), k-block = 64.
#define PROW 72    // padded LDS row stride for 64-wide P tile
__global__ void __launch_bounds__(128)
mha_attn(const _Float16* __restrict__ Qf, const _Float16* __restrict__ Kf,
         const _Float16* __restrict__ Vtf, _Float16* __restrict__ Yf) {
    __shared__ __align__(32) _Float16 psm[4][16 * PROW];

    const int wave = threadIdx.x >> 5;
    const int lane = threadIdx.x & 31;
    const int hl   = lane >> 4;
    const int lr   = lane & 15;
    const int bh   = blockIdx.x;
    const int q0   = blockIdx.y * 64 + wave * 16;

    const _Float16* Qb = Qf  + (size_t)bh * TT * DH;
    const _Float16* Kb = Kf  + (size_t)bh * TT * DH;
    const _Float16* Vb = Vtf + (size_t)bh * DH * TT;

    // Q A-fragments: d = 0..31 and 32..63 (scale folded in at QKV store)
    const _Float16* qrow = Qb + (size_t)(q0 + lr) * DH;
    U16H qa0, qa1;
    qa0.h[0] = *(const v8h*)(qrow + 16 * hl);
    qa0.h[1] = *(const v8h*)(qrow + 8 + 16 * hl);
    qa1.h[0] = *(const v8h*)(qrow + 32 + 16 * hl);
    qa1.h[1] = *(const v8h*)(qrow + 32 + 8 + 16 * hl);

    v8f O[4] = {};                 // 16q x 64d accumulator
    float mrun[8], lrun[8];
#pragma unroll
    for (int r = 0; r < 8; ++r) { mrun[r] = -1e30f; lrun[r] = 0.0f; }

    _Float16* myp = &psm[wave][0];

    for (int kb = 0; kb < TT; kb += 64) {
        // ----- S = Q * K^T  (16q x 64k) -----
        v8f S[4] = {};
#pragma unroll
        for (int jj = 0; jj < 4; ++jj) {
            const _Float16* krow = Kb + (size_t)(kb + 16 * jj + lr) * DH;
            v16h blo = *(const v16h*)(krow + 16 * hl);        // d 0..31
            v16h bhi = *(const v16h*)(krow + 32 + 16 * hl);   // d 32..63
            S[jj] = wmma_f16(qa0.v, blo, S[jj]);
            S[jj] = wmma_f16(qa1.v, bhi, S[jj]);
        }

        // ----- online softmax (one shuffle tree per 64 columns) -----
        float alpha[8];
#pragma unroll
        for (int r = 0; r < 8; ++r) {
            float mb = fmaxf(fmaxf(S[0][r], S[1][r]), fmaxf(S[2][r], S[3][r]));
            mb = fmaxf(mb, __shfl_xor(mb, 1, 32));
            mb = fmaxf(mb, __shfl_xor(mb, 2, 32));
            mb = fmaxf(mb, __shfl_xor(mb, 4, 32));
            mb = fmaxf(mb, __shfl_xor(mb, 8, 32));
            float mnew = fmaxf(mrun[r], mb);
            alpha[r]   = __expf(mrun[r] - mnew);
            mrun[r]    = mnew;
            float p0 = __expf(S[0][r] - mnew);
            float p1 = __expf(S[1][r] - mnew);
            float p2 = __expf(S[2][r] - mnew);
            float p3 = __expf(S[3][r] - mnew);
            S[0][r] = p0; S[1][r] = p1; S[2][r] = p2; S[3][r] = p3;
            float rs = (p0 + p1) + (p2 + p3);
            rs += __shfl_xor(rs, 1, 32);
            rs += __shfl_xor(rs, 2, 32);
            rs += __shfl_xor(rs, 4, 32);
            rs += __shfl_xor(rs, 8, 32);
            lrun[r] = lrun[r] * alpha[r] + rs;
        }
#pragma unroll
        for (int r = 0; r < 8; ++r) {
            O[0][r] *= alpha[r]; O[1][r] *= alpha[r];
            O[2][r] *= alpha[r]; O[3][r] *= alpha[r];
        }

        // ----- C-layout P -> LDS -> two A-layout fragments (k 0..31,32..63) -
#pragma unroll
        for (int r = 0; r < 8; ++r) {
            int row = r + 8 * hl;
#pragma unroll
            for (int jj = 0; jj < 4; ++jj)
                myp[row * PROW + 16 * jj + lr] = (_Float16)S[jj][r];
        }
        U16H pa0, pa1;   // per-wave private region; DS ops in-order per wave
        pa0.h[0] = *(const v8h*)(myp + lr * PROW + 16 * hl);
        pa0.h[1] = *(const v8h*)(myp + lr * PROW + 8 + 16 * hl);
        pa1.h[0] = *(const v8h*)(myp + lr * PROW + 32 + 16 * hl);
        pa1.h[1] = *(const v8h*)(myp + lr * PROW + 40 + 16 * hl);

        // ----- O += P * V  (V stored [d][t]: contiguous B loads) -----
#pragma unroll
        for (int j = 0; j < 4; ++j) {
            const _Float16* vrow = Vb + (size_t)(16 * j + lr) * TT + kb;
            v16h vb0 = *(const v16h*)(vrow + 16 * hl);        // k 0..31
            v16h vb1 = *(const v16h*)(vrow + 32 + 16 * hl);   // k 32..63
            O[j] = wmma_f16(pa0.v, vb0, O[j]);
            O[j] = wmma_f16(pa1.v, vb1, O[j]);
        }
    }

    // ----- normalize and store f16 into [B*T, DIM] for the proj GEMM -----
    const int bb = bh >> 4;
    const int hh = bh & 15;
#pragma unroll
    for (int r = 0; r < 8; ++r) {
        float inv = 1.0f / lrun[r];
        int q = q0 + r + 8 * hl;
        size_t base = ((size_t)(bb * TT + q)) * DIM + hh * DH;
        Yf[base + lr]      = (_Float16)(O[0][r] * inv);
        Yf[base + 16 + lr] = (_Float16)(O[1][r] * inv);
        Yf[base + 32 + lr] = (_Float16)(O[2][r] * inv);
        Yf[base + 48 + lr] = (_Float16)(O[3][r] * inv);
    }
}

// ------------------------------- launcher ----------------------------------

extern "C" void kernel_launch(void* const* d_in, const int* in_sizes, int n_in,
                              void* d_out, int out_size, void* d_ws, size_t ws_size,
                              hipStream_t stream) {
    const float* x     = (const float*)d_in[0];   // [2,2048,1024]
    const float* Wqkv  = (const float*)d_in[1];   // [1024,3072]
    const float* bqkv  = (const float*)d_in[2];   // [3072]
    const float* Wproj = (const float*)d_in[3];   // [1024,1024]
    const float* bproj = (const float*)d_in[4];   // [1024]
    float* out = (float*)d_out;

    _Float16* ws = (_Float16*)d_ws;
    const size_t NX    = (size_t)MTOT * DIM;            // 4M halves
    const size_t NWQKV = (size_t)DIM * 3 * DIM;         // 3M
    const size_t NWPRJ = (size_t)DIM * DIM;             // 1M
    const size_t NQKV1 = (size_t)BBATCH * NH * TT * DH; // 4M each
    _Float16* x16    = ws;
    _Float16* wqkvt  = x16 + NX;
    _Float16* wprojt = wqkvt + NWQKV;
    _Float16* Qf     = wprojt + NWPRJ;
    _Float16* Kf     = Qf + NQKV1;
    _Float16* Vtf    = Kf + NQKV1;
    _Float16* attn16 = Vtf + NQKV1;

    // 1) precision / layout prep
    mha_cvt_f16<<<4096, 256, 0, stream>>>(x, x16, (int)NX);
    mha_cvt_t_f16<<<4096, 256, 0, stream>>>(Wqkv, wqkvt, DIM, 3 * DIM);
    mha_cvt_t_f16<<<2048, 256, 0, stream>>>(Wproj, wprojt, DIM, DIM);

    // 2) QKV GEMM + head scatter (Q pre-scaled, V transposed)
    mha_qkv_gemm<<<dim3(MTOT / 128, (3 * DIM) / 64), 128, 0, stream>>>(
        x16, wqkvt, bqkv, Qf, Kf, Vtf);

    // 3) flash attention
    mha_attn<<<dim3(BBATCH * NH, TT / 64), 128, 0, stream>>>(Qf, Kf, Vtf, attn16);

    // 4) output projection
    mha_proj_gemm<<<dim3(MTOT / 128, DIM / 64), 128, 0, stream>>>(
        attn16, wprojt, bproj, out);
}
